// Gpt2Attention_19207093748056
// MI455X (gfx1250) — matrix operand order repaired and checked by reading, not compiled, not run on hardware
//
#include <hip/hip_runtime.h>
#include <stdint.h>
#include <stddef.h>

// ---------------------------------------------------------------------------
// Types for CDNA5 WMMA (gfx1250, wave32)
// ---------------------------------------------------------------------------
typedef __bf16 bf16_t;
typedef __attribute__((ext_vector_type(16))) __bf16 v16bf;
typedef __attribute__((ext_vector_type(8)))  float    v8f;
typedef __attribute__((ext_vector_type(4)))  uint32_t u32x4;
// TDM descriptor groups (probe-verified builtin operand types)
typedef __attribute__((ext_vector_type(4))) unsigned int tdm_g0_t;
typedef __attribute__((ext_vector_type(8))) int          tdm_g1_t;
typedef __attribute__((ext_vector_type(4))) int          tdm_g2_t;

#define WMMA_BF16(a, b, c) \
  __builtin_amdgcn_wmma_f32_16x16x32_bf16(false, (a), false, (b), (short)0, (c), false, false)

// fp32 -> bf16 round-to-nearest-even
__device__ __forceinline__ bf16_t f2bf(float f) {
  uint32_t u = __builtin_bit_cast(uint32_t, f);
  uint32_t r = u + 0x7FFFu + ((u >> 16) & 1u);
  uint16_t h = (uint16_t)(r >> 16);
  return __builtin_bit_cast(bf16_t, h);
}

struct FragBits { u32x4 lo; u32x4 hi; };
__device__ __forceinline__ v16bf pack_frag(u32x4 lo, u32x4 hi) {
  FragBits t; t.lo = lo; t.hi = hi;
  return __builtin_bit_cast(v16bf, t);
}

// ---------------------------------------------------------------------------
// Tensor Data Mover: async 2D tile load global -> LDS (D# per ISA 08 §8).
// data_size = 2 bytes. dim0 = contiguous axis. Issued by one wave only;
// completion tracked with TENSORcnt.
// ---------------------------------------------------------------------------
__device__ __forceinline__ void tdm_load_2d(unsigned lds_byte_off,
                                            const void* gaddr,
                                            unsigned tensor_d0, unsigned tensor_d1,
                                            unsigned tile_d0, unsigned tile_d1,
                                            unsigned long long stride0_elems) {
  unsigned long long ga = (unsigned long long)(uintptr_t)gaddr;
  tdm_g0_t g0 = {0u, 0u, 0u, 0u};
  g0[0] = 1u;                                    // count=1, user mode
  g0[1] = lds_byte_off;                          // lds_addr
  g0[2] = (unsigned)(ga & 0xFFFFFFFFull);        // global_addr[31:0]
  g0[3] = (unsigned)((ga >> 32) & 0x01FFFFFFull) // global_addr[56:32]
          | (2u << 30);                          // type = 2 ("image")
  tdm_g1_t g1 = {0, 0, 0, 0, 0, 0, 0, 0};
  g1[0] = (int)(1u << 16);                       // data_size=1 -> 2 bytes
  g1[1] = (int)((tensor_d0 & 0xFFFFu) << 16);    // tensor_dim0[15:0]
  g1[2] = (int)(((tensor_d0 >> 16) & 0xFFFFu) | ((tensor_d1 & 0xFFFFu) << 16));
  g1[3] = (int)(((tensor_d1 >> 16) & 0xFFFFu) | (tile_d0 << 16));  // tile_dim0
  g1[4] = (int)tile_d1;                          // tile_dim1 (tile_dim2 = 0)
  g1[5] = (int)(stride0_elems & 0xFFFFFFFFull);  // tensor_dim0_stride[31:0]
  g1[6] = (int)((stride0_elems >> 32) & 0xFFFFull);
  g1[7] = 0;
  tdm_g2_t z4 = {0, 0, 0, 0};
#if __clang_major__ >= 23
  tdm_g1_t z8 = {0, 0, 0, 0, 0, 0, 0, 0};
  __builtin_amdgcn_tensor_load_to_lds(g0, g1, z4, z4, z8, 0);
#else
  __builtin_amdgcn_tensor_load_to_lds(g0, g1, z4, z4, 0);
#endif
}

// ---------------------------------------------------------------------------
// f32 -> bf16 conversion (bandwidth-trivial prologue)
// ---------------------------------------------------------------------------
__global__ void cvt_f32_bf16(const float* __restrict__ in, bf16_t* __restrict__ out, int n) {
  int i = blockIdx.x * blockDim.x + threadIdx.x;
  int stride = gridDim.x * blockDim.x;
  for (; i < n; i += stride) out[i] = f2bf(in[i]);
}

// ---------------------------------------------------------------------------
// Tiled bf16 WMMA GEMM with TDM-staged, double-buffered LDS tiles.
// C[M,N] = A[M,K] @ B[K,N] + bias[N]
// Block: 128 threads = 4 waves (2x2); block tile 128x128; wave tile 64x64;
// K-step 32. Dynamic LDS: 2 x (A 128x32 + B 32x128) bf16 = 32 KB, so TDM
// lds_addr offsets are 0-based.
// ---------------------------------------------------------------------------
template <bool OUT_BF16>
__global__ __launch_bounds__(128, 1) void gemm128_bf16(
    const bf16_t* __restrict__ A, const bf16_t* __restrict__ Bm,
    const float* __restrict__ bias, void* __restrict__ Cout,
    int M, int N, int K) {
  extern __shared__ bf16_t smem[];  // [2][A:4096 | B:4096] elements

  const int tid  = threadIdx.x;
  const int lane = tid & 31;
  const int wid  = tid >> 5;
  const int wm   = (wid >> 1) * 64;
  const int wn   = (wid & 1) * 64;
  const int m0   = blockIdx.y * 128;
  const int n0   = blockIdx.x * 128;
  const int lm   = lane & 15;
  const int lh   = lane >> 4;

  v8f acc[4][4];
#pragma unroll
  for (int i = 0; i < 4; i++)
#pragma unroll
    for (int j = 0; j < 4; j++)
#pragma unroll
      for (int e = 0; e < 8; e++) acc[i][j][e] = 0.0f;

  // Prologue: async-stage first K tile into buffer 0 (wave 0 issues DMA)
  if (wid == 0) {
    tdm_load_2d(/*lds*/ 0u, A + (size_t)m0 * K, (unsigned)K, (unsigned)M,
                32u, 128u, (unsigned long long)K);
    tdm_load_2d(/*lds*/ 8192u, Bm + n0, (unsigned)N, (unsigned)K,
                128u, 32u, (unsigned long long)N);
  }

  int p = 0;
  for (int k0 = 0; k0 < K; k0 += 32) {
    if (wid == 0) __builtin_amdgcn_s_wait_tensorcnt(0);
    __syncthreads();  // tile p ready for everyone

    // Prefetch next K tile into the other buffer while we compute
    int kn = k0 + 32;
    if (kn < K && wid == 0) {
      unsigned lds = (unsigned)((p ^ 1) * 16384);
      tdm_load_2d(lds, A + (size_t)m0 * K + kn, (unsigned)K, (unsigned)M,
                  32u, 128u, (unsigned long long)K);
      tdm_load_2d(lds + 8192u, Bm + (size_t)kn * N + n0, (unsigned)N, (unsigned)K,
                  128u, 32u, (unsigned long long)N);
    }

    const bf16_t* lA = smem + p * 8192;         // [128][32]
    const bf16_t* lB = smem + p * 8192 + 4096;  // [32][128]

    // A fragments: lane=M, K pairs -> two contiguous 16B chunks
    v16bf afr[4];
#pragma unroll
    for (int mt = 0; mt < 4; mt++) {
      int row = wm + mt * 16 + lm;
      int kh  = lh * 16;
      u32x4 lo = *(const u32x4*)(lA + row * 32 + kh);
      u32x4 hi = *(const u32x4*)(lA + row * 32 + kh + 8);
      afr[mt] = pack_frag(lo, hi);
    }
    // B fragments: lane%16=N, lane/16 selects K half; gather K pairs
    v16bf bfr[4];
#pragma unroll
    for (int nt = 0; nt < 4; nt++) {
      int col = wn + nt * 16 + lm;
      int kh  = lh * 16;
      v16bf b;
#pragma unroll
      for (int v = 0; v < 8; v++) {
        b[2 * v]     = lB[(kh + 2 * v) * 128 + col];
        b[2 * v + 1] = lB[(kh + 2 * v + 1) * 128 + col];
      }
      bfr[nt] = b;
    }

#pragma unroll
    for (int mt = 0; mt < 4; mt++)
#pragma unroll
      for (int nt = 0; nt < 4; nt++)
        acc[mt][nt] = WMMA_BF16(afr[mt], bfr[nt], acc[mt][nt]);

    __syncthreads();  // all reads of buffer p done before it is refilled
    p ^= 1;
  }

  // Epilogue: C layout lane%16=N, VGPR v -> M = v + 8*(lane/16)
#pragma unroll
  for (int mt = 0; mt < 4; mt++) {
#pragma unroll
    for (int nt = 0; nt < 4; nt++) {
      int colb = n0 + wn + nt * 16 + lm;
      float bv = bias[colb];
#pragma unroll
      for (int v = 0; v < 8; v++) {
        int row = m0 + wm + mt * 16 + v + 8 * lh;
        float val = acc[mt][nt][v] + bv;
        if (OUT_BF16)
          ((bf16_t*)Cout)[(size_t)row * N + colb] = f2bf(val);
        else
          ((float*)Cout)[(size_t)row * N + colb] = val;
      }
    }
  }
}

// ---------------------------------------------------------------------------
// Causal flash attention over qkv [B*S, 3*D] bf16, heads H=16, HD=64.
// Grid: (S/64, B*H). Block: 128 threads = 4 waves; each wave owns 16 q rows.
// Q and K^T fragments load directly from global (HD contiguous). V tiles are
// TDM-staged ([key][hd] in LDS), double-buffered, overlapping the QK^T WMMAs.
// Dynamic LDS: 2 x V(64x64) + P(4 waves x 16x64) bf16 = 24 KB.
// ---------------------------------------------------------------------------
__global__ __launch_bounds__(128, 1) void flash_attn64(
    const bf16_t* __restrict__ qkv, bf16_t* __restrict__ outp) {
  constexpr int S_ = 2048, Dm = 1024, T3 = 3072, HDc = 64;
  extern __shared__ bf16_t smem[];  // V0 @0, V1 @4096, P @8192 (elements)

  const int tid  = threadIdx.x;
  const int lane = tid & 31;
  const int wid  = tid >> 5;
  const int lm   = lane & 15;
  const int lh   = lane >> 4;

  const int bh    = blockIdx.y;
  const int b     = bh >> 4;
  const int h     = bh & 15;
  const int qbase = blockIdx.x * 64;
  const int wq    = qbase + wid * 16;

  const bf16_t* qp = qkv + (size_t)(b * S_) * T3 + h * HDc;
  const bf16_t* kp = qp + Dm;
  const bf16_t* vp = qp + 2 * Dm;

  // Persistent Q fragments (A layout, K-dim = hd)
  v16bf aq[2];
  {
    const bf16_t* qr = qp + (size_t)(wq + lm) * T3;
    int kh8 = lh * 8;
    aq[0] = pack_frag(*(const u32x4*)(qr + 2 * kh8),      *(const u32x4*)(qr + 2 * kh8 + 8));
    aq[1] = pack_frag(*(const u32x4*)(qr + 32 + 2 * kh8), *(const u32x4*)(qr + 40 + 2 * kh8));
  }

  v8f o[4];
#pragma unroll
  for (int nt = 0; nt < 4; nt++)
#pragma unroll
    for (int v = 0; v < 8; v++) o[nt][v] = 0.0f;

  float mrow[8], lrow[8];
#pragma unroll
  for (int v = 0; v < 8; v++) { mrow[v] = -3.0e38f; lrow[v] = 0.0f; }

  const float scale = 0.125f;        // 1/sqrt(64)
  const int nkb = (qbase >> 6) + 1;  // causal key-block count

  // Prologue: async-stage V block 0
  if (wid == 0)
    tdm_load_2d(0u, vp, (unsigned)T3, (unsigned)S_, 64u, 64u,
                (unsigned long long)T3);

  int p = 0;
  for (int kb = 0; kb < nkb; kb++) {
    const int kbase = kb * 64;
    if (wid == 0) __builtin_amdgcn_s_wait_tensorcnt(0);
    __syncthreads();  // V buffer p ready

    if (kb + 1 < nkb && wid == 0)
      tdm_load_2d((unsigned)((p ^ 1) * 8192),
                  vp + (size_t)(kbase + 64) * T3,
                  (unsigned)T3, (unsigned)S_, 64u, 64u,
                  (unsigned long long)T3);

    // ---- Load ALL K^T B-fragments first (distinct regs -> one long load
    //      clause, WMMAs then drain the loads with decreasing loadcnt) ----
    v16bf bk[4][2];
#pragma unroll
    for (int kt = 0; kt < 4; kt++) {
      const bf16_t* kr = kp + (size_t)(kbase + kt * 16 + lm) * T3;
      int kh16 = lh * 16;
      bk[kt][0] = pack_frag(*(const u32x4*)(kr + kh16),
                            *(const u32x4*)(kr + kh16 + 8));
      bk[kt][1] = pack_frag(*(const u32x4*)(kr + 32 + kh16),
                            *(const u32x4*)(kr + 32 + kh16 + 8));
    }

    // ---- Scores S = Q K^T : 4 col tiles x 2 K-depth steps = 8 WMMA ----
    v8f sc[4];
#pragma unroll
    for (int kt = 0; kt < 4; kt++)
#pragma unroll
      for (int v = 0; v < 8; v++) sc[kt][v] = 0.0f;
#pragma unroll
    for (int kt = 0; kt < 4; kt++) {
      sc[kt] = WMMA_BF16(aq[0], bk[kt][0], sc[kt]);
      sc[kt] = WMMA_BF16(aq[1], bk[kt][1], sc[kt]);
    }

    // ---- Scale + causal mask ----
#pragma unroll
    for (int kt = 0; kt < 4; kt++) {
      int col = kbase + kt * 16 + lm;
#pragma unroll
      for (int v = 0; v < 8; v++) {
        int row = wq + v + 8 * lh;
        float s = sc[kt][v] * scale;
        sc[kt][v] = (col <= row) ? s : -3.0e38f;
      }
    }

    // ---- Online softmax (row reductions across the 16-lane N group) ----
#pragma unroll
    for (int v = 0; v < 8; v++) {
      float mx = fmaxf(fmaxf(sc[0][v], sc[1][v]), fmaxf(sc[2][v], sc[3][v]));
#pragma unroll
      for (int off = 1; off < 16; off <<= 1)
        mx = fmaxf(mx, __shfl_xor(mx, off, 32));
      float mnew  = fmaxf(mrow[v], mx);
      float alpha = __expf(mrow[v] - mnew);
      mrow[v] = mnew;
      float ls = 0.0f;
#pragma unroll
      for (int kt = 0; kt < 4; kt++) {
        float pv = __expf(sc[kt][v] - mnew);
        sc[kt][v] = pv;
        ls += pv;
      }
#pragma unroll
      for (int off = 1; off < 16; off <<= 1)
        ls += __shfl_xor(ls, off, 32);
      lrow[v] = lrow[v] * alpha + ls;
#pragma unroll
      for (int nt = 0; nt < 4; nt++) o[nt][v] *= alpha;
    }

    // ---- P (C layout) -> per-wave LDS -> A-layout fragments ----
    bf16_t* pw = smem + 8192 + wid * (16 * 64);
#pragma unroll
    for (int kt = 0; kt < 4; kt++)
#pragma unroll
      for (int v = 0; v < 8; v++)
        pw[(v + 8 * lh) * 64 + kt * 16 + lm] = f2bf(sc[kt][v]);

    v16bf ap[2];
    {
      const bf16_t* pr = pw + lm * 64;
      int kh8 = lh * 8;
      ap[0] = pack_frag(*(const u32x4*)(pr + 2 * kh8),      *(const u32x4*)(pr + 2 * kh8 + 8));
      ap[1] = pack_frag(*(const u32x4*)(pr + 32 + 2 * kh8), *(const u32x4*)(pr + 40 + 2 * kh8));
    }

    // ---- O += P V : V is [key][hd] in LDS; gather B fragments ----
    const bf16_t* Vl = smem + p * 4096;
#pragma unroll
    for (int nt = 0; nt < 4; nt++) {
      int n    = nt * 16 + lm;
      int kh16 = lh * 16;
      v16bf bv0, bv1;
#pragma unroll
      for (int v = 0; v < 8; v++) {
        bv0[2 * v]     = Vl[(kh16 + 2 * v) * 64 + n];
        bv0[2 * v + 1] = Vl[(kh16 + 2 * v + 1) * 64 + n];
        bv1[2 * v]     = Vl[(32 + kh16 + 2 * v) * 64 + n];
        bv1[2 * v + 1] = Vl[(32 + kh16 + 2 * v + 1) * 64 + n];
      }
      o[nt] = WMMA_BF16(ap[0], bv0, o[nt]);
      o[nt] = WMMA_BF16(ap[1], bv1, o[nt]);
    }

    __syncthreads();  // all reads of V buffer p done before refill
    p ^= 1;
  }

  // Normalize and write attention output [B*S, D] bf16
  float inv[8];
#pragma unroll
  for (int v = 0; v < 8; v++) inv[v] = 1.0f / lrow[v];
  bf16_t* orow = outp + (size_t)(b * S_) * Dm + h * HDc;
#pragma unroll
  for (int nt = 0; nt < 4; nt++) {
#pragma unroll
    for (int v = 0; v < 8; v++) {
      int row = wq + v + 8 * lh;
      orow[(size_t)row * Dm + nt * 16 + lm] = f2bf(o[nt][v] * inv[v]);
    }
  }
}

// ---------------------------------------------------------------------------
// Host launcher
// ---------------------------------------------------------------------------
extern "C" void kernel_launch(void* const* d_in, const int* in_sizes, int n_in,
                              void* d_out, int out_size, void* d_ws, size_t ws_size,
                              hipStream_t stream) {
  (void)in_sizes; (void)n_in; (void)out_size; (void)ws_size;

  const float* x      = (const float*)d_in[0];
  const float* w_attn = (const float*)d_in[1];
  const float* b_attn = (const float*)d_in[2];
  const float* w_proj = (const float*)d_in[3];
  const float* b_proj = (const float*)d_in[4];
  float* out = (float*)d_out;

  constexpr int Bn = 2, Sn = 2048, Dn = 1024, Hn = 16;
  constexpr int Mrows = Bn * Sn;                      // 4096
  constexpr size_t n_x   = (size_t)Mrows * Dn;
  constexpr size_t n_wa  = (size_t)Dn * 3 * Dn;
  constexpr size_t n_wp  = (size_t)Dn * Dn;
  constexpr size_t n_qkv = (size_t)Mrows * 3 * Dn;

  // Workspace carve-out (bf16): ~50.3 MB total
  bf16_t* xbf  = (bf16_t*)d_ws;
  bf16_t* wabf = xbf + n_x;
  bf16_t* wpbf = wabf + n_wa;
  bf16_t* qkv  = wpbf + n_wp;
  bf16_t* aout = qkv + n_qkv;

  // 1) f32 -> bf16 conversions
  cvt_f32_bf16<<<dim3(2048), dim3(256), 0, stream>>>(x, xbf, (int)n_x);
  cvt_f32_bf16<<<dim3(2048), dim3(256), 0, stream>>>(w_attn, wabf, (int)n_wa);
  cvt_f32_bf16<<<dim3(1024), dim3(256), 0, stream>>>(w_proj, wpbf, (int)n_wp);

  // 2) QKV projection: [4096,1024] @ [1024,3072] + b_attn -> bf16 qkv
  gemm128_bf16<true><<<dim3(3072 / 128, Mrows / 128), dim3(128), 32768, stream>>>(
      xbf, wabf, b_attn, (void*)qkv, Mrows, 3 * Dn, Dn);

  // 3) Causal flash attention -> bf16 [4096,1024]
  flash_attn64<<<dim3(Sn / 64, Bn * Hn), dim3(128), 24576, stream>>>(qkv, aout);

  // 4) Output projection: [4096,1024] @ [1024,1024] + b_proj -> f32 out
  gemm128_bf16<false><<<dim3(Dn / 128, Mrows / 128), dim3(128), 32768, stream>>>(
      aout, wpbf, b_proj, (void*)out, Mrows, Dn, Dn);
}
